// SAE_V_16312285790429
// MI455X (gfx1250) — matrix operand order repaired and checked by reading, not compiled, not run on hardware
//
#include <hip/hip_runtime.h>

#define BATCH 4096
#define DIM   1024
#define HID   16384
#define RANK  33   // threshold = 33rd-largest (top_vals[:, 32]); keep strictly greater

typedef __attribute__((ext_vector_type(16))) __bf16 v16bf;
typedef __attribute__((ext_vector_type(8)))  float  v8f;

union BfVec { unsigned short u[16]; uint4 q[2]; v16bf v; };

__device__ __forceinline__ unsigned short f32_to_bf16_rne(float f) {
    unsigned int x = __float_as_uint(f);
    x += 0x7FFFu + ((x >> 16) & 1u);     // round-to-nearest-even
    return (unsigned short)(x >> 16);
}
__device__ __forceinline__ float bf16_to_f32(unsigned short h) {
    return __uint_as_float(((unsigned int)h) << 16);
}

// ---------------------------------------------------------------------------
// Kernel 0: one-time split-conversion of enc_w (f32) into WMMA-fragment-
// swizzled bf16 hi/lo planes. Fragment (n_tile, k_tile) = 32 lanes x 16 bf16
// contiguous (1 KB), so the encoder fetches a whole B fragment per lane with
// two b128 loads. 64 MB total -> L2-resident (192 MB L2).
// B-fragment element e of lane l maps to K = k_tile*32 + e + 16*(l>>4),
// N = n_tile*16 + (l&15).
// ---------------------------------------------------------------------------
__launch_bounds__(256)
__global__ void sae_convert_encw(const float* __restrict__ enc_w,        // [DIM][HID]
                                 unsigned short* __restrict__ whi,       // [HID/16][DIM/32][32][16]
                                 unsigned short* __restrict__ wlo)
{
    const int gid    = blockIdx.x * 256 + threadIdx.x;   // one (fragment, lane) pair
    const int lane   = gid & 31;
    const int frag   = gid >> 5;                         // 0 .. (HID/16)*(DIM/32)-1
    const int k_tile = frag & (DIM / 32 - 1);
    const int n_tile = frag >> 5;                        // DIM/32 == 32
    const int half   = lane >> 4;
    const int n      = n_tile * 16 + (lane & 15);
    const size_t base = (size_t)frag * 512 + (size_t)lane * 16;

    BfVec h, l;
#pragma unroll
    for (int e = 0; e < 16; ++e) {
        const int k = k_tile * 32 + e + 16 * half;
        const float w = enc_w[(size_t)k * HID + n];
        const unsigned short hb = f32_to_bf16_rne(w);
        h.u[e] = hb;
        l.u[e] = f32_to_bf16_rne(w - bf16_to_f32(hb));
    }
    *reinterpret_cast<uint4*>(whi + base)     = h.q[0];
    *reinterpret_cast<uint4*>(whi + base + 8) = h.q[1];
    *reinterpret_cast<uint4*>(wlo + base)     = l.q[0];
    *reinterpret_cast<uint4*>(wlo + base + 8) = l.q[1];
}

// ---------------------------------------------------------------------------
// Shared helper: cooperative split-bf16 staging of the 16 x DIM A panel.
// ---------------------------------------------------------------------------
__device__ __forceinline__ void stage_A(const float* __restrict__ x, int m0, int tid,
                                        unsigned short (*a_hi)[DIM],
                                        unsigned short (*a_lo)[DIM])
{
    for (int i = tid; i < 16 * (DIM / 4); i += 256) {
        const int m = i >> 8;               // / (DIM/4)
        const int k = (i & 255) << 2;
        const float4 f = *reinterpret_cast<const float4*>(x + (size_t)(m0 + m) * DIM + k);
        const float fv[4] = {f.x, f.y, f.z, f.w};
#pragma unroll
        for (int j = 0; j < 4; ++j) {
            const unsigned short h = f32_to_bf16_rne(fv[j]);
            a_hi[m][k + j] = h;
            a_lo[m][k + j] = f32_to_bf16_rne(fv[j] - bf16_to_f32(h));
        }
    }
}

__device__ __forceinline__ void store_latent(const v8f& acc, const float* __restrict__ enc_b,
                                             float* __restrict__ lat, int m0, int col, int half)
{
    const float bias = enc_b[col];
#pragma unroll
    for (int r = 0; r < 8; ++r) {
        const int row = m0 + r + 8 * half;   // C/D layout: lanes 16-31 -> M = r+8
        float v = acc[r] + bias;
        v = (v > 0.f) ? v : 0.f;             // relu
        lat[(size_t)row * HID + col] = v;
    }
}

// ---------------------------------------------------------------------------
// Kernel 1 (fast path): encoder GEMM with pre-converted split-bf16 weights.
// latent = relu(x @ enc_w + enc_b); a*b ~= ah*bh + ah*bl + al*bh.
// Block = 8 waves; tile = 16 rows x 128 cols; one 16x16 N-tile per wave.
// Inner loop: 4x ds_load_b128 (A) + 4x global b128 (B) + 3 WMMAs.
// ---------------------------------------------------------------------------
__launch_bounds__(256)
__global__ void sae_encode_wmma_pre(const float* __restrict__ xv,
                                    const float* __restrict__ xt,
                                    const unsigned short* __restrict__ whi,
                                    const unsigned short* __restrict__ wlo,
                                    const float* __restrict__ enc_b,
                                    float* __restrict__ lat_base)      // [2][BATCH][HID]
{
    __shared__ unsigned short a_hi[16][DIM];   // 32 KB
    __shared__ unsigned short a_lo[16][DIM];   // 32 KB

    const int tid = threadIdx.x;
    const int z   = blockIdx.z;
    const int m0  = blockIdx.y * 16;
    stage_A((z == 0) ? xv : xt, m0, tid, a_hi, a_lo);
    __syncthreads();

    const int wave = tid >> 5;
    const int lane = tid & 31;
    const int half = lane >> 4;
    const int lrow = lane & 15;
    const int n_tile = blockIdx.x * 8 + wave;

    v8f acc = {};
    size_t fbase = ((size_t)n_tile * (DIM / 32)) * 512 + (size_t)lane * 16;

    for (int kb = 0; kb < DIM; kb += 32, fbase += 512) {
        // A fragments from LDS (one contiguous 32B run per plane per lane).
        BfVec ah, al;
#pragma unroll
        for (int e = 0; e < 16; ++e) {
            const int ka = kb + e + 16 * half;
            ah.u[e] = a_hi[lrow][ka];
            al.u[e] = a_lo[lrow][ka];
        }
        // B fragments: contiguous 32B per lane, b128-vectorized, L2-served.
        BfVec bh, bl;
        bh.q[0] = *reinterpret_cast<const uint4*>(whi + fbase);
        bh.q[1] = *reinterpret_cast<const uint4*>(whi + fbase + 8);
        bl.q[0] = *reinterpret_cast<const uint4*>(wlo + fbase);
        bl.q[1] = *reinterpret_cast<const uint4*>(wlo + fbase + 8);

        if (kb + 32 < DIM)
            __builtin_prefetch(whi + fbase + 512, 0, 1);   // global_prefetch_b8

        acc = __builtin_amdgcn_wmma_f32_16x16x32_bf16(false, ah.v, false, bh.v, (short)0, acc, false, false);
        acc = __builtin_amdgcn_wmma_f32_16x16x32_bf16(false, ah.v, false, bl.v, (short)0, acc, false, false);
        acc = __builtin_amdgcn_wmma_f32_16x16x32_bf16(false, al.v, false, bh.v, (short)0, acc, false, false);
    }

    store_latent(acc, enc_b, lat_base + (size_t)z * BATCH * HID,
                 m0, n_tile * 16 + lrow, half);
}

// ---------------------------------------------------------------------------
// Kernel 1 (fallback, small-ws path): inline weight conversion (previous round).
// ---------------------------------------------------------------------------
__launch_bounds__(256)
__global__ void sae_encode_wmma_inline(const float* __restrict__ xv,
                                       const float* __restrict__ xt,
                                       const float* __restrict__ enc_w,
                                       const float* __restrict__ enc_b,
                                       float* __restrict__ lat_base)
{
    __shared__ unsigned short a_hi[16][DIM];
    __shared__ unsigned short a_lo[16][DIM];

    const int tid = threadIdx.x;
    const int z   = blockIdx.z;
    const int m0  = blockIdx.y * 16;
    stage_A((z == 0) ? xv : xt, m0, tid, a_hi, a_lo);
    __syncthreads();

    const int wave = tid >> 5;
    const int lane = tid & 31;
    const int half = lane >> 4;
    const int lrow = lane & 15;
    const int n0   = blockIdx.x * 128 + wave * 16;

    v8f acc = {};
    for (int kb = 0; kb < DIM; kb += 32) {
        BfVec ah, al;
#pragma unroll
        for (int e = 0; e < 16; ++e) {
            const int ka = kb + e + 16 * half;
            ah.u[e] = a_hi[lrow][ka];
            al.u[e] = a_lo[lrow][ka];
        }
        BfVec bh, bl;
#pragma unroll
        for (int e = 0; e < 16; ++e) {
            const float w = enc_w[(size_t)(kb + e + 16 * half) * HID + n0 + lrow];
            const unsigned short h = f32_to_bf16_rne(w);
            bh.u[e] = h;
            bl.u[e] = f32_to_bf16_rne(w - bf16_to_f32(h));
        }
        acc = __builtin_amdgcn_wmma_f32_16x16x32_bf16(false, ah.v, false, bh.v, (short)0, acc, false, false);
        acc = __builtin_amdgcn_wmma_f32_16x16x32_bf16(false, ah.v, false, bl.v, (short)0, acc, false, false);
        acc = __builtin_amdgcn_wmma_f32_16x16x32_bf16(false, al.v, false, bh.v, (short)0, acc, false, false);
    }

    store_latent(acc, enc_b, lat_base + (size_t)z * BATCH * HID, m0, n0 + lrow, half);
}

// ---------------------------------------------------------------------------
// Kernel 2: per-row exact top-k threshold (byte-wise radix select over IEEE
// bits; post-relu values >=0 so uint order == float order), in-place sparsify,
// then sparse decode recon = sum_{kept h} latent[h] * dec_w[h,:] + dec_b.
// One 256-thread block per row; row resident in registers (64 f32/lane).
// ---------------------------------------------------------------------------
#define FOR_EACH_VAL(EXPR)                                                     \
    _Pragma("unroll") for (int j = 0; j < 16; ++j) {                           \
        { float v = v4[j].x; int h = hbase + 4 * j + 0; EXPR; }                \
        { float v = v4[j].y; int h = hbase + 4 * j + 1; EXPR; }                \
        { float v = v4[j].z; int h = hbase + 4 * j + 2; EXPR; }                \
        { float v = v4[j].w; int h = hbase + 4 * j + 3; EXPR; }                \
    }

__launch_bounds__(256)
__global__ void sae_topk_decode(float* __restrict__ lat_base,      // [2][BATCH][HID]
                                const float* __restrict__ dec_w,   // [HID][DIM]
                                const float* __restrict__ dec_b,   // [DIM]
                                float* __restrict__ recon_base)    // [2][BATCH][DIM]
{
    __shared__ unsigned int hist[256];
    __shared__ unsigned int s_prefix;
    __shared__ int s_rank, s_nkept;
    __shared__ int   keep_idx[64];
    __shared__ float keep_val[64];

    const int tid = threadIdx.x;
    const int row = blockIdx.x;
    const int z   = blockIdx.y;
    float* lat = lat_base + (size_t)z * BATCH * HID + (size_t)row * HID;

    const int hbase = tid * 64;
    float4 v4[16];
#pragma unroll
    for (int j = 0; j < 16; ++j)
        v4[j] = reinterpret_cast<const float4*>(lat)[tid * 16 + j];

    if (tid == 0) { s_prefix = 0u; s_rank = RANK; s_nkept = 0; }
    __syncthreads();

    for (int shift = 24; shift >= 0; shift -= 8) {
        hist[tid] = 0u;
        __syncthreads();
        const unsigned int hiMask = (shift == 24) ? 0u : (0xFFFFFFFFu << (shift + 8));
        const unsigned int pfx = s_prefix;
        FOR_EACH_VAL({
            (void)h;
            const unsigned int u = __float_as_uint(v);
            if ((u & hiMask) == pfx) atomicAdd(&hist[(u >> shift) & 255u], 1u);
        });
        __syncthreads();
        if (tid == 0) {
            int rank = s_rank, cum = 0, bin = 0;
            for (int b = 255; b >= 0; --b) {
                const int c = (int)hist[b];
                if (cum + c >= rank) { bin = b; s_rank = rank - cum; break; }
                cum += c;
            }
            s_prefix = pfx | ((unsigned int)bin << shift);
        }
        __syncthreads();
    }
    const float thres = __uint_as_float(s_prefix);   // exact 33rd-largest value

    FOR_EACH_VAL({
        const bool keep = v > thres;
        lat[h] = keep ? v : 0.f;
        if (keep) {
            const int p = atomicAdd(&s_nkept, 1);
            if (p < 64) { keep_idx[p] = h; keep_val[p] = v; }
        }
    });
    __syncthreads();
    int nk = s_nkept; if (nk > 64) nk = 64;

    float4 acc = reinterpret_cast<const float4*>(dec_b)[tid];
    for (int j = 0; j < nk; ++j) {
        const float  s = keep_val[j];
        const float4 w = reinterpret_cast<const float4*>(dec_w + (size_t)keep_idx[j] * DIM)[tid];
        acc.x += s * w.x; acc.y += s * w.y; acc.z += s * w.z; acc.w += s * w.w;
    }
    float* recon = recon_base + (size_t)z * BATCH * DIM + (size_t)row * DIM;
    reinterpret_cast<float4*>(recon)[tid] = acc;
}

// ---------------------------------------------------------------------------
// d_out layout (flat, return order): recon_v [B*D], recon_t [B*D],
//                                    latent_v [B*H], latent_t [B*H]
// ---------------------------------------------------------------------------
extern "C" void kernel_launch(void* const* d_in, const int* in_sizes, int n_in,
                              void* d_out, int out_size, void* d_ws, size_t ws_size,
                              hipStream_t stream) {
    (void)in_sizes; (void)n_in; (void)out_size;
    const float* xv    = (const float*)d_in[0];
    const float* xt    = (const float*)d_in[1];
    const float* enc_w = (const float*)d_in[2];
    const float* enc_b = (const float*)d_in[3];
    const float* dec_w = (const float*)d_in[4];
    const float* dec_b = (const float*)d_in[5];

    float* out        = (float*)d_out;
    float* recon_base = out;                              // [2][BATCH][DIM]
    float* lat_base   = out + (size_t)2 * BATCH * DIM;    // [2][BATCH][HID]

    const size_t plane = (size_t)DIM * HID;               // bf16 elements per plane
    const size_t need  = plane * 2 * sizeof(unsigned short);  // 64 MB

    dim3 g1(HID / 128, BATCH / 16, 2);
    if (ws_size >= need) {
        unsigned short* whi = (unsigned short*)d_ws;
        unsigned short* wlo = whi + plane;
        const int frags = (HID / 16) * (DIM / 32);        // 32768
        sae_convert_encw<<<frags * 32 / 256, 256, 0, stream>>>(enc_w, whi, wlo);
        sae_encode_wmma_pre<<<g1, 256, 0, stream>>>(xv, xt, whi, wlo, enc_b, lat_base);
    } else {
        sae_encode_wmma_inline<<<g1, 256, 0, stream>>>(xv, xt, enc_w, enc_b, lat_base);
    }

    dim3 g2(BATCH, 2);
    sae_topk_decode<<<g2, 256, 0, stream>>>(lat_base, dec_w, dec_b, recon_base);
}
